// FP8Linear_20555713479186
// MI455X (gfx1250) — matrix operand order repaired and checked by reading, not compiled, not run on hardware
//
#include <hip/hip_runtime.h>

// ---------------------------------------------------------------------------
// FP8(E4M3)-fake-quant linear: out[M,N] = x[M,K] @ qw^T[N,K] + bias
//   M = 8192, K = 4096, N = 4096.
// qw/scale is a multiple of 1/8, |.| <= 240 -> EXACT in fp16 (Q, workspace).
// x (fp32) is Dekker-split x = hi + lo (both fp16); two
// v_wmma_f32_16x16x32_f16 accumulations per tile give fp32-grade accuracy.
// Double-buffered LDS pipeline (1 barrier/k-step), fragment-swizzled LDS
// layout (conflict-free ds_load_b128), async global->LDS (ASYNCcnt) for B.
// ---------------------------------------------------------------------------

typedef _Float16 half_t;
typedef __attribute__((ext_vector_type(16))) _Float16 v16h;
typedef __attribute__((ext_vector_type(8)))  _Float16 v8h;
typedef __attribute__((ext_vector_type(4)))  _Float16 v4h;
typedef __attribute__((ext_vector_type(8)))  float    v8f;
typedef __attribute__((ext_vector_type(4)))  float    v4f;
typedef __attribute__((ext_vector_type(4)))  int      v4i;

#define K_DIM   4096
#define N_DIM   4096
#define M_DIM   8192
#define E4M3MAX 240.0f

#define BM 128
#define BN 128
#define BK 32
// fragment-swizzled tile: [tile16][lane32][16 halves] = 8 KB per 128x32 tile
#define TILE_H (8 * 32 * 16)

#if defined(__has_builtin)
#if __has_builtin(__builtin_amdgcn_global_load_async_to_lds_b128)
#define HAVE_ASYNC_LDS 1
#endif
#endif
#ifndef HAVE_ASYNC_LDS
#define HAVE_ASYNC_LDS 0
#endif

// ---------------- workspace layout ----------------
// byte 0   : unsigned amax bits
// byte 4   : float scale
// byte 256 : half_t Q[N_DIM*K_DIM]  (exact fp16 quantized weight, 32 MiB)

__global__ void fp8lin_init(unsigned* __restrict__ amax) {
    if (blockIdx.x == 0 && threadIdx.x == 0) amax[0] = 0u;
}

__global__ void fp8lin_amax(const float* __restrict__ w, unsigned* __restrict__ amax, int n) {
    float m = 0.0f;
    for (int i = blockIdx.x * blockDim.x + threadIdx.x; i < n; i += gridDim.x * blockDim.x)
        m = fmaxf(m, fabsf(w[i]));
    __shared__ float red[256];
    red[threadIdx.x] = m;
    __syncthreads();
    for (int s = 128; s > 0; s >>= 1) {
        if (threadIdx.x < s) red[threadIdx.x] = fmaxf(red[threadIdx.x], red[threadIdx.x + s]);
        __syncthreads();
    }
    if (threadIdx.x == 0) atomicMax(amax, __float_as_uint(red[0]));  // vals >= 0
}

__global__ void fp8lin_scale(const unsigned* __restrict__ amax, float* __restrict__ scale) {
    if (blockIdx.x == 0 && threadIdx.x == 0) {
        float a = __uint_as_float(amax[0]);
        scale[0] = fmaxf(a / E4M3MAX, 1e-10f);
    }
}

__global__ void fp8lin_quant(const float* __restrict__ w,
                             const float* __restrict__ scale_p,
                             half_t* __restrict__ q, int n4) {
    int i = blockIdx.x * blockDim.x + threadIdx.x;
    if (i >= n4) return;
    const float s = scale_p[0];
    const v4f wv = ((const v4f*)w)[i];
    v4h out;
#pragma unroll
    for (int e = 0; e < 4; ++e) {
        float t = wv[e] / s;
        t = fminf(fmaxf(t, -E4M3MAX), E4M3MAX);
        t = (float)(_Float16)t;            // fp16 RNE round-trip
        t = rintf(t * 8.0f) * 0.125f;      // mantissa truncation (RNE)
        t = fminf(fmaxf(t, -E4M3MAX), E4M3MAX);
        out[e] = (half_t)t;                // exact
    }
    ((v4h*)q)[i] = out;
}

static __device__ inline v16h pack16(v8h a, v8h b) {
    return __builtin_shufflevector(a, b, 0, 1, 2, 3, 4, 5, 6, 7,
                                         8, 9, 10, 11, 12, 13, 14, 15);
}

static __device__ inline void wait_async0() {
#if HAVE_ASYNC_LDS
#if __has_builtin(__builtin_amdgcn_s_wait_asynccnt)
    __builtin_amdgcn_s_wait_asynccnt(0);
#else
    asm volatile("s_wait_asynccnt 0x0" ::: "memory");
#endif
#endif
}

// ---- A tile: 128x32 fp32 -> hi/lo fp16, stored in WMMA-fragment order.
// A frag as staged here (same k order as B): lane chunk c (halves 8c..8c+7) holds K group g=2*laneHi+c.
static __device__ inline void load_A_regs(const float* __restrict__ x, int m0, int kt,
                                          int tid, v4f* aReg) {
#pragma unroll
    for (int i = 0; i < 4; ++i) {
        const int idx = tid + i * 256;          // 0..1023
        const int r = idx >> 3, c4 = (idx & 7) * 4;
        aReg[i] = *(const v4f*)(x + (size_t)(m0 + r) * K_DIM + kt + c4);
    }
}

static __device__ inline void cvt_store_A(half_t* __restrict__ dAh, half_t* __restrict__ dAl,
                                          int tid, const v4f* aReg) {
#pragma unroll
    for (int i = 0; i < 4; ++i) {
        const int idx = tid + i * 256;
        const int r = idx >> 3, c4 = (idx & 7) * 4;
        const int mt = r >> 4, g = c4 >> 3;            // K group of 8 halves
        const int lane = (r & 15) + ((g >> 1) << 4);   // laneHi = g>>1
        const int h0 = (mt * 32 + lane) * 16 + ((g & 1) << 3) + (c4 & 7);
        v4h h, l;
#pragma unroll
        for (int e = 0; e < 4; ++e) {
            const float v = aReg[i][e];
            h[e] = (half_t)v;
            l[e] = (half_t)(v - (float)h[e]);
        }
        *(v4h*)(dAh + h0) = h;
        *(v4h*)(dAl + h0) = l;
    }
}

// ---- B tile: 128x32 fp16, fragment order. B frag as staged here (k order matches A): lane holds
// contiguous K halves 16*laneHi..+15 of column n=laneLo -> contiguous in Q.
#if HAVE_ASYNC_LDS
static __device__ inline void async_copy16(const half_t* g, half_t* l) {
    // signature (from hipcc diagnostic): (int4 AS1*, int4 AS3*, imm, imm)
    __builtin_amdgcn_global_load_async_to_lds_b128(
        (__attribute__((address_space(1))) v4i*)(v4i*)g,
        (__attribute__((address_space(3))) v4i*)(v4i*)l,
        0, 0);
}
static __device__ inline void issue_B_async(const half_t* __restrict__ Q, int n0, int kt,
                                            int tid, half_t* __restrict__ dB) {
#pragma unroll
    for (int i = 0; i < 2; ++i) {
        const int idx = tid + i * 256;          // 0..511
        const int r = idx >> 2, c8 = (idx & 3) * 8;
        const int t = r >> 4;
        const int lane = (r & 15) + (c8 & 16);
        const int h0 = (t * 32 + lane) * 16 + (c8 & 8);
        async_copy16(Q + (size_t)(n0 + r) * K_DIM + kt + c8, dB + h0);
    }
}
#else
static __device__ inline void load_B_regs(const half_t* __restrict__ Q, int n0, int kt,
                                          int tid, v8h* bReg) {
#pragma unroll
    for (int i = 0; i < 2; ++i) {
        const int idx = tid + i * 256;
        const int r = idx >> 2, c8 = (idx & 3) * 8;
        bReg[i] = *(const v8h*)(Q + (size_t)(n0 + r) * K_DIM + kt + c8);
    }
}
static __device__ inline void store_B_lds(half_t* __restrict__ dB, int tid, const v8h* bReg) {
#pragma unroll
    for (int i = 0; i < 2; ++i) {
        const int idx = tid + i * 256;
        const int r = idx >> 2, c8 = (idx & 3) * 8;
        const int t = r >> 4;
        const int lane = (r & 15) + (c8 & 16);
        const int h0 = (t * 32 + lane) * 16 + (c8 & 8);
        *(v8h*)(dB + h0) = bReg[i];
    }
}
#endif

__global__ __launch_bounds__(256) void fp8lin_gemm(
    const float* __restrict__ x,      // [M_DIM, K_DIM]
    const half_t* __restrict__ Q,     // [N_DIM, K_DIM] (B column-major)
    const float* __restrict__ scale_p,
    const float* __restrict__ bias,   // [N_DIM]
    float* __restrict__ out)          // [M_DIM, N_DIM]
{
    __shared__ __align__(16) half_t sAh[2][TILE_H];
    __shared__ __align__(16) half_t sAl[2][TILE_H];
    __shared__ __align__(16) half_t sB [2][TILE_H];   // 48 KB total

    const int tid    = threadIdx.x;
    const int lane   = tid & 31;
    const int w      = tid >> 5;
    const int laneLo = lane & 15;
    const int laneHi = lane >> 4;

    const int m0 = blockIdx.y * BM;
    const int n0 = blockIdx.x * BN;
    const int wm = (w & 3) * 32;        // wave: 32 rows
    const int wn = (w >> 2) * 64;       // wave: 64 cols

    v8f acc[2][4];
#pragma unroll
    for (int mi = 0; mi < 2; ++mi)
#pragma unroll
        for (int ni = 0; ni < 4; ++ni)
            acc[mi][ni] = (v8f){0.f, 0.f, 0.f, 0.f, 0.f, 0.f, 0.f, 0.f};

    // ---- prologue: stage tile kt=0 into buffer 0
    v4f aReg[4];
#if !HAVE_ASYNC_LDS
    v8h bReg[2];
#endif
    load_A_regs(x, m0, 0, tid, aReg);
#if HAVE_ASYNC_LDS
    issue_B_async(Q, n0, 0, tid, sB[0]);
#else
    load_B_regs(Q, n0, 0, tid, bReg);
    store_B_lds(sB[0], tid, bReg);
#endif
    cvt_store_A(sAh[0], sAl[0], tid, aReg);
    wait_async0();
    __syncthreads();

    int buf = 0;
    for (int kt = 0; kt < K_DIM; kt += BK) {
        const int nxt = buf ^ 1;
        const int ktn = kt + BK;
        const bool more = ktn < K_DIM;

        // ---- prefetch next tile (overlaps with compute below)
        if (more) {
#if HAVE_ASYNC_LDS
            issue_B_async(Q, n0, ktn, tid, sB[nxt]);
#else
            load_B_regs(Q, n0, ktn, tid, bReg);
#endif
            load_A_regs(x, m0, ktn, tid, aReg);
        }

        // ---- fragments: each lane reads its contiguous 32B (conflict-free)
        const half_t* pAh = sAh[buf];
        const half_t* pAl = sAl[buf];
        const half_t* pB  = sB[buf];
        v16h afh[2], afl[2], bf[4];
#pragma unroll
        for (int mi = 0; mi < 2; ++mi) {
            const int base = (((wm >> 4) + mi) * 32 + lane) * 16;
            afh[mi] = pack16(*(const v8h*)(pAh + base), *(const v8h*)(pAh + base + 8));
            afl[mi] = pack16(*(const v8h*)(pAl + base), *(const v8h*)(pAl + base + 8));
        }
#pragma unroll
        for (int ni = 0; ni < 4; ++ni) {
            const int base = (((wn >> 4) + ni) * 32 + lane) * 16;
            bf[ni] = pack16(*(const v8h*)(pB + base), *(const v8h*)(pB + base + 8));
        }

        // ---- 16 WMMAs: lo then hi into the same f32 accumulator
#pragma unroll
        for (int mi = 0; mi < 2; ++mi)
#pragma unroll
            for (int ni = 0; ni < 4; ++ni) {
                acc[mi][ni] = __builtin_amdgcn_wmma_f32_16x16x32_f16(
                    false, afl[mi], false, bf[ni], (short)0, acc[mi][ni], false, false);
                acc[mi][ni] = __builtin_amdgcn_wmma_f32_16x16x32_f16(
                    false, afh[mi], false, bf[ni], (short)0, acc[mi][ni], false, false);
            }

        // ---- drain prefetch into the other buffer
        if (more) {
            cvt_store_A(sAh[nxt], sAl[nxt], tid, aReg);
#if !HAVE_ASYNC_LDS
            store_B_lds(sB[nxt], tid, bReg);
#endif
        }
        wait_async0();
        __syncthreads();
        buf = nxt;
    }

    // ---- epilogue: out = scale*acc + bias   (C/D: VGPR r -> M=r+8*laneHi, N=laneLo)
    const float s = scale_p[0];
#pragma unroll
    for (int mi = 0; mi < 2; ++mi)
#pragma unroll
        for (int ni = 0; ni < 4; ++ni) {
            const int n  = n0 + wn + ni * 16 + laneLo;
            const float bv = bias[n];
#pragma unroll
            for (int r = 0; r < 8; ++r) {
                const int m = m0 + wm + mi * 16 + laneHi * 8 + r;
                out[(size_t)m * N_DIM + n] = fmaf(s, acc[mi][ni][r], bv);
            }
        }
}

extern "C" void kernel_launch(void* const* d_in, const int* in_sizes, int n_in,
                              void* d_out, int out_size, void* d_ws, size_t ws_size,
                              hipStream_t stream) {
    const float* x      = (const float*)d_in[0];   // [4,2048,4096]
    const float* weight = (const float*)d_in[1];   // [4096,4096]
    const float* bias   = (const float*)d_in[2];   // [4096]
    float* out          = (float*)d_out;

    unsigned* amax = (unsigned*)d_ws;
    float*    scl  = (float*)((char*)d_ws + 4);
    half_t*   Q    = (half_t*)((char*)d_ws + 256); // 32 MiB

    const int NW = N_DIM * K_DIM;

    fp8lin_init<<<1, 64, 0, stream>>>(amax);
    fp8lin_amax<<<2048, 256, 0, stream>>>(weight, amax, NW);
    fp8lin_scale<<<1, 64, 0, stream>>>(amax, scl);
    fp8lin_quant<<<NW / 4 / 256, 256, 0, stream>>>(weight, scl, Q, NW / 4);

    dim3 grid(N_DIM / BN, M_DIM / BM);   // 32 x 64
    fp8lin_gemm<<<grid, 256, 0, stream>>>(x, Q, scl, bias, out);
    (void)in_sizes; (void)n_in; (void)out_size; (void)ws_size;
}
